// SE3Convolution_11888469475374
// MI455X (gfx1250) — matrix operand order repaired and checked by reading, not compiled, not run on hardware
//
#include <hip/hip_runtime.h>
#include <stdint.h>

typedef __attribute__((ext_vector_type(16))) __bf16          v16bf;
typedef __attribute__((ext_vector_type(8)))  float           v8f;
typedef __attribute__((ext_vector_type(16))) unsigned short  us16;

#define S      5
#define NTAP   125          // 5*5*5
#define NCH    52
#define CPAD   64
#define IN_D   48
#define OUT_D  44
#define NB     8
#define TILE_N 48           // one ow tile covers all 44 outputs (3 x 16 subtiles)
#define NWAVES 12           // 4 mchunks x 3 nsub
#define NTHR   (NWAVES * 32)
#define WIN    52           // max col span: kw(0..4) + nn(0..47)
#define TILE_ELEMS (WIN * CPAD)            // 3328
#define REGS_PER_T ((TILE_ELEMS + NTHR - 1) / NTHR)   // 9
#define AFRAG_HALVES ((size_t)NTAP * 2 * 4 * 32 * 16)  // 512000 per table

__device__ __forceinline__ unsigned short f32_to_bf16_rne(float f) {
    unsigned u = __float_as_uint(f);
    unsigned r = u + 0x7FFFu + ((u >> 16) & 1u);
    return (unsigned short)(r >> 16);
}
__device__ __forceinline__ float bf16_to_f32(unsigned short h) {
    return __uint_as_float(((unsigned)h) << 16);
}

struct BasisPtrs { const float* b[9]; };

// ---------------------------------------------------------------------------
// Stage 1: build W[co][ci][tap] from weight+bases, split into bf16 hi/lo, and
// scatter directly into WMMA A-fragment layout:
//   idx = (((tap*2 + kc)*4 + mchunk)*32 + lane)*16 + h
// A 16-bit 16x32 layout (contiguous, matches B): lane<16 -> K0..15,
//                                               lane>=16 -> K16..31.
// ---------------------------------------------------------------------------
__global__ void build_afrag(const float* __restrict__ weight, BasisPtrs bp,
                            unsigned short* __restrict__ Ahi,
                            unsigned short* __restrict__ Alo) {
    int gid = blockIdx.x * blockDim.x + threadIdx.x;
    if (gid >= NTAP * CPAD * CPAD) return;
    int tap = gid / (CPAD * CPAD);
    int rem = gid % (CPAD * CPAD);
    int co  = rem / CPAD;
    int ci  = rem % CPAD;

    float w = 0.0f;
    if (co < NCH && ci < NCH) {
        int i, u, a;
        if      (co < 8)  { i = 0; u = co;          a = 0; }
        else if (co < 32) { i = 1; int r = co - 8;  u = r / 3; a = r % 3; }
        else              { i = 2; int r = co - 32; u = r / 5; a = r % 5; }
        int j, v, b;
        if      (ci < 8)  { j = 0; v = ci;          b = 0; }
        else if (ci < 32) { j = 1; int r = ci - 8;  v = r / 3; b = r % 3; }
        else              { j = 2; int r = ci - 32; v = r / 5; b = r % 5; }

        const int woff[3][3] = {{0, 256, 512}, {640, 896, 1152}, {1280, 1408, 1536}};
        const int mults[3] = {8, 8, 4};
        const int dims [3] = {1, 3, 5};

        const float* bas = bp.b[i * 3 + j];
        int mi  = mults[j];
        int dov = dims[i];     // out-irrep dim (a)
        int djv = dims[j];     // in-irrep dim  (b)
        int wbase = woff[i][j] + (u * mi + v) * 4;
        float acc = 0.0f;
        #pragma unroll
        for (int k = 0; k < 4; ++k) {
            float bw = bas[(((size_t)(k * dov + a) * djv + b) * NTAP) + tap];
            acc += weight[wbase + k] * bw;
        }
        w = acc;
    }

    unsigned short hi = f32_to_bf16_rne(w);
    unsigned short lo = f32_to_bf16_rne(w - bf16_to_f32(hi));

    int mchunk = co >> 4, mlane = co & 15;
    int kc = ci >> 5,     kk    = ci & 31;
    int lane, h;
    if      (kk < 16) { lane = mlane;      h = kk;      }
    else              { lane = mlane + 16; h = kk - 16; }
    size_t idx = ((((size_t)tap * 2 + kc) * 4 + mchunk) * 32 + lane) * 16 + h;
    Ahi[idx] = hi;
    Alo[idx] = lo;
}

// ---------------------------------------------------------------------------
// Staging helpers: global fp32 -> registers, then registers -> LDS as bf16
// hi/lo in [col][ci] layout (so a lane's B fragment is one 32B contiguous read)
// ---------------------------------------------------------------------------
__device__ __forceinline__ void load_tile(const float* __restrict__ x,
                                          int n, int iz, int iy, int tid,
                                          float regs[REGS_PER_T]) {
    #pragma unroll
    for (int k = 0; k < REGS_PER_T; ++k) {
        int idx = tid + k * NTHR;
        float f = 0.0f;
        if (idx < TILE_ELEMS) {
            int xx = idx >> 6;
            int ci = idx & 63;
            if (ci < NCH) {
                int col = xx;
                if (col > IN_D - 1) col = IN_D - 1;  // clamped cols feed only discarded N-pad
                f = x[((((size_t)n * NCH + ci) * IN_D + iz) * IN_D + iy) * IN_D + col];
            }
        }
        regs[k] = f;
    }
}

__device__ __forceinline__ void store_tile(unsigned short* __restrict__ dhi,
                                           unsigned short* __restrict__ dlo,
                                           int tid, const float regs[REGS_PER_T]) {
    #pragma unroll
    for (int k = 0; k < REGS_PER_T; ++k) {
        int idx = tid + k * NTHR;
        if (idx < TILE_ELEMS) {
            unsigned short h = f32_to_bf16_rne(regs[k]);
            dhi[idx] = h;
            dlo[idx] = f32_to_bf16_rne(regs[k] - bf16_to_f32(h));
        }
    }
}

// ---------------------------------------------------------------------------
// Stage 2: implicit-GEMM 3D conv. Block = 64(Cout pad) x 48(ow pad) tile at
// fixed (n, od, oh). 12 waves = 4 mchunks x 3 n-subtiles, one v8f acc each.
// Software-pipelined over the 25 (kd,kh) slices with double-buffered LDS:
// global loads for slice t+1 are issued before the WMMA loop on slice t and
// converted/stored to the alternate buffer after it, hiding load latency
// behind 360 WMMAs per slice.
// ---------------------------------------------------------------------------
__global__ void __launch_bounds__(NTHR)
se3conv_wmma(const float* __restrict__ x,
             const unsigned short* __restrict__ Ahi,
             const unsigned short* __restrict__ Alo,
             float* __restrict__ out) {
    __shared__ alignas(32) unsigned short sh_hi[2][TILE_ELEMS];
    __shared__ alignas(32) unsigned short sh_lo[2][TILE_ELEMS];

    const int tid    = threadIdx.x;
    const int wave   = tid >> 5;
    const int lane   = tid & 31;
    const int mchunk = wave & 3;
    const int nsub   = wave >> 2;          // 0..2

    const int oh = blockIdx.x;             // 0..43
    const int od = blockIdx.y;             // 0..43
    const int n  = blockIdx.z;             // 0..7

    const int nn = (lane & 15) + nsub * 16;      // local N: 0..47
    const int koff_lane = (lane < 16) ? 0 : 16;  // B layout: lane<16 -> K0..15

    v8f acc = {};
    float regs[REGS_PER_T];

    // prologue: stage slice 0
    load_tile(x, n, od, oh, tid, regs);
    store_tile(sh_hi[0], sh_lo[0], tid, regs);
    __syncthreads();

    for (int t = 0; t < S * S; ++t) {            // t = kd*5 + kh
        const int cur = t & 1;

        // issue global loads for next slice before compute (latency hidden)
        if (t + 1 < S * S) {
            int kd = (t + 1) / S;
            int kh = (t + 1) % S;
            load_tile(x, n, od + kd, oh + kh, tid, regs);
        }

        const unsigned short* chi = sh_hi[cur];
        const unsigned short* clo = sh_lo[cur];

        #pragma unroll
        for (int kw = 0; kw < S; ++kw) {
            const int tap = t * S + kw;
            const int xx  = kw + nn;
            #pragma unroll
            for (int kc = 0; kc < 2; ++kc) {
                size_t fidx = (((size_t)tap * 2 + kc) * 4 + mchunk) * 32 + lane;
                us16 araw_hi = *(const us16*)(Ahi + fidx * 16);
                us16 araw_lo = *(const us16*)(Alo + fidx * 16);
                const unsigned short* bph = chi + xx * CPAD + kc * 32 + koff_lane;
                const unsigned short* bpl = clo + xx * CPAD + kc * 32 + koff_lane;
                us16 braw_hi = *(const us16*)bph;
                us16 braw_lo = *(const us16*)bpl;

                v16bf ah = __builtin_bit_cast(v16bf, araw_hi);
                v16bf al = __builtin_bit_cast(v16bf, araw_lo);
                v16bf bh = __builtin_bit_cast(v16bf, braw_hi);
                v16bf bl = __builtin_bit_cast(v16bf, braw_lo);

                acc = __builtin_amdgcn_wmma_f32_16x16x32_bf16(
                          false, ah, false, bh, (short)0, acc, false, false);
                acc = __builtin_amdgcn_wmma_f32_16x16x32_bf16(
                          false, ah, false, bl, (short)0, acc, false, false);
                acc = __builtin_amdgcn_wmma_f32_16x16x32_bf16(
                          false, al, false, bh, (short)0, acc, false, false);
            }
        }

        // convert + store next slice into the other buffer, then barrier
        if (t + 1 < S * S) {
            store_tile(sh_hi[cur ^ 1], sh_lo[cur ^ 1], tid, regs);
        }
        __syncthreads();
    }

    // C/D layout: VGPR r -> M = r + (lane<16 ? 0 : 8), N = lane&15
    const int ow = nn;
    if (ow < OUT_D) {
        const int mbase = mchunk * 16 + ((lane < 16) ? 0 : 8);
        #pragma unroll
        for (int r = 0; r < 8; ++r) {
            int co = mbase + r;
            if (co < NCH) {
                out[((((size_t)n * NCH + co) * OUT_D + od) * OUT_D + oh) * OUT_D + ow] = acc[r];
            }
        }
    }
}

extern "C" void kernel_launch(void* const* d_in, const int* in_sizes, int n_in,
                              void* d_out, int out_size, void* d_ws, size_t ws_size,
                              hipStream_t stream) {
    const float* x      = (const float*)d_in[0];
    const float* weight = (const float*)d_in[1];
    BasisPtrs bp;
    for (int i = 0; i < 9; ++i) bp.b[i] = (const float*)d_in[2 + i];

    unsigned short* Ahi = (unsigned short*)d_ws;
    unsigned short* Alo = Ahi + AFRAG_HALVES;
    float* out = (float*)d_out;

    int total = NTAP * CPAD * CPAD;
    hipLaunchKernelGGL(build_afrag, dim3((total + 255) / 256), dim3(256), 0, stream,
                       weight, bp, Ahi, Alo);

    dim3 grid(OUT_D, OUT_D, NB);
    hipLaunchKernelGGL(se3conv_wmma, grid, dim3(NTHR), 0, stream, x, Ahi, Alo, out);
}
